// RmiNet_23751169147101
// MI455X (gfx1250) — matrix operand order repaired and checked by reading, not compiled, not run on hardware
//
#include <hip/hip_runtime.h>
#include <hip/hip_bf16.h>
#include <math.h>

#define EPSBN 1e-5f

typedef __attribute__((ext_vector_type(16))) _Float16 v16h;
typedef __attribute__((ext_vector_type(8)))  float    v8f;

__device__ __forceinline__ float gelu_f(float x) {
  // exact (erf-based) GELU, matching jax.nn.gelu(approximate=False)
  return 0.5f * x * (1.0f + erff(x * 0.70710678118654752440f));
}

// ---------------------------------------------------------------------------
// Kernel A: fused conv1+bn1+gelu  ->  conv2(dil3)+bn2+gelu  ->  hflat[B,1024]
// One workgroup (256 threads, 8 waves) per batch row. x row staged in LDS,
// BN folded into conv weights held in VGPRs, b128 LDS traffic throughout.
// ---------------------------------------------------------------------------
__global__ __launch_bounds__(256) void conv_fused_kernel(
    const float* __restrict__ x,
    const float* __restrict__ c1w, const float* __restrict__ c1b,
    const float* __restrict__ g1, const float* __restrict__ b1,
    const float* __restrict__ m1, const float* __restrict__ v1,
    const float* __restrict__ c2w, const float* __restrict__ c2b,
    const float* __restrict__ g2, const float* __restrict__ b2,
    const float* __restrict__ m2, const float* __restrict__ v2,
    float* __restrict__ hflat)
{
  __shared__ __align__(16) float sX[6 * 1032];   // x padded by 4 each side
  __shared__ __align__(16) float sG[6 * 1036];   // gelu(bn(conv1)) padded by 4 each side
  __shared__ float sWf1[90], sB1[6], sWf2[30], sB2s[1];

  const int tid = threadIdx.x;
  const int b   = blockIdx.x;

  // ---- fold BN params into conv weights (few threads; everyone else loads x)
  if (tid < 90) {
    const int co = tid / 30;
    const float s = g1[co] * rsqrtf(v1[co] + EPSBN);
    sWf1[tid] = c1w[tid] * s;
  } else if (tid < 96) {
    const int co = tid - 90;
    const float s = g1[co] * rsqrtf(v1[co] + EPSBN);
    sB1[co] = (c1b[co] - m1[co]) * s + b1[co];
  } else if (tid < 126) {
    const float s = g2[0] * rsqrtf(v2[0] + EPSBN);
    sWf2[tid - 96] = c2w[tid - 96] * s;
  } else if (tid == 126) {
    const float s = g2[0] * rsqrtf(v2[0] + EPSBN);
    sB2s[0] = (c2b[0] - m2[0]) * s + b2[0];
  } else if (tid >= 128 && tid < 176) {        // zero x pads (6 ch * 8 pads)
    const int i = tid - 128, ci = i >> 3, j = i & 7;
    sX[ci * 1032 + ((j < 4) ? j : (1028 + j - 4))] = 0.f;
  } else if (tid >= 176 && tid < 224) {        // zero g pads
    const int i = tid - 176, ci = i >> 3, j = i & 7;
    sG[ci * 1036 + ((j < 4) ? j : (1032 + j - 4))] = 0.f;
  }

  // ---- stream the 24 KB x row into LDS with b128 ops
  const float4* xv = reinterpret_cast<const float4*>(x + (size_t)b * 6144);
  for (int i = tid; i < 1536; i += 256) {
    const int ci = i >> 8, t4 = i & 255;
    float4 u = xv[i];
    *reinterpret_cast<float4*>(&sX[ci * 1032 + 4 + t4 * 4]) = u;
  }
  __syncthreads();

  // ---- hoist folded weights into VGPRs (constant-indexed -> stays in regs)
  float wf1r[90], b1r[6], wf2r[30];
#pragma unroll
  for (int i = 0; i < 90; ++i) wf1r[i] = sWf1[i];
#pragma unroll
  for (int i = 0; i < 6; ++i) b1r[i] = sB1[i];
#pragma unroll
  for (int i = 0; i < 30; ++i) wf2r[i] = sWf2[i];
  const float b2r = sB2s[0];

  // ---- conv1 + bn1 + gelu : each thread computes 4 consecutive t1 positions
  {
    const int t1 = tid * 4;                    // 0..1020
    float xr[6][8];
#pragma unroll
    for (int ci = 0; ci < 6; ++ci) {
      float4 u = *reinterpret_cast<const float4*>(&sX[ci * 1032 + t1]);
      float4 w = *reinterpret_cast<const float4*>(&sX[ci * 1032 + t1 + 4]);
      xr[ci][0] = u.x; xr[ci][1] = u.y; xr[ci][2] = u.z; xr[ci][3] = u.w;
      xr[ci][4] = w.x; xr[ci][5] = w.y; xr[ci][6] = w.z; xr[ci][7] = w.w;
    }
#pragma unroll
    for (int co = 0; co < 6; ++co) {
      float gq[4];
#pragma unroll
      for (int j = 0; j < 4; ++j) {
        float acc = b1r[co];
#pragma unroll
        for (int ci = 0; ci < 6; ++ci)
#pragma unroll
          for (int k = 0; k < 5; ++k)
            acc = fmaf(wf1r[co * 30 + ci * 5 + k], xr[ci][j + k], acc);
        gq[j] = gelu_f(acc);
      }
      float4 gv; gv.x = gq[0]; gv.y = gq[1]; gv.z = gq[2]; gv.w = gq[3];
      *reinterpret_cast<float4*>(&sG[co * 1036 + 4 + t1]) = gv;
    }
    if (tid < 4) {                             // tail: t1 = 1024..1027
      const int t = 1024 + tid;
#pragma unroll
      for (int co = 0; co < 6; ++co) {
        float acc = b1r[co];
#pragma unroll
        for (int ci = 0; ci < 6; ++ci)
#pragma unroll
          for (int k = 0; k < 5; ++k)
            acc = fmaf(wf1r[co * 30 + ci * 5 + k], sX[ci * 1032 + t + k], acc);
        sG[co * 1036 + 4 + t] = gelu_f(acc);
      }
    }
  }
  __syncthreads();

  // ---- conv2 (dilation 3) + bn2 + gelu -> hflat ; 4 outputs per thread
  {
    const int t2 = tid * 4;                    // exactly covers 0..1023
    float gr[6][16];
#pragma unroll
    for (int ci = 0; ci < 6; ++ci)
#pragma unroll
      for (int q = 0; q < 4; ++q) {
        float4 u = *reinterpret_cast<const float4*>(&sG[ci * 1036 + t2 + q * 4]);
        gr[ci][q * 4 + 0] = u.x; gr[ci][q * 4 + 1] = u.y;
        gr[ci][q * 4 + 2] = u.z; gr[ci][q * 4 + 3] = u.w;
      }
    float o4[4];
#pragma unroll
    for (int j = 0; j < 4; ++j) {
      float acc = b2r;
#pragma unroll
      for (int ci = 0; ci < 6; ++ci)
#pragma unroll
        for (int k = 0; k < 5; ++k)
          acc = fmaf(wf2r[ci * 5 + k], gr[ci][3 * k + j], acc);
      o4[j] = gelu_f(acc);
    }
    float4 ov; ov.x = o4[0]; ov.y = o4[1]; ov.z = o4[2]; ov.w = o4[3];
    *reinterpret_cast<float4*>(&hflat[(size_t)b * 1024 + t2]) = ov;
  }
}

// ---------------------------------------------------------------------------
// 3x3 closest-rotation (== U diag(1,1,det(U V^T)) V^T from SVD).
// Jacobi eigensolver on A^T A -> V, sigma ; U = A V / sigma ; s = sign(det A).
// ---------------------------------------------------------------------------
template <int P, int Q>
__device__ __forceinline__ void jrot(float (&S)[3][3], float (&V)[3][3]) {
  const float apq = S[P][Q];
  if (fabsf(apq) > 1e-24f) {
    const float tau = (S[Q][Q] - S[P][P]) / (2.0f * apq);
    const float t = copysignf(1.0f, tau) / (fabsf(tau) + sqrtf(1.0f + tau * tau));
    const float c = rsqrtf(1.0f + t * t);
    const float s = t * c;
#pragma unroll
    for (int k = 0; k < 3; ++k) { float a = S[k][P], bq = S[k][Q]; S[k][P] = c * a - s * bq; S[k][Q] = s * a + c * bq; }
#pragma unroll
    for (int k = 0; k < 3; ++k) { float a = S[P][k], bq = S[Q][k]; S[P][k] = c * a - s * bq; S[Q][k] = s * a + c * bq; }
#pragma unroll
    for (int k = 0; k < 3; ++k) { float a = V[k][P], bq = V[k][Q]; V[k][P] = c * a - s * bq; V[k][Q] = s * a + c * bq; }
  }
}

__device__ __forceinline__ void csw(float& a, float& b) { float t = a; a = b; b = t; }

__device__ __forceinline__ void rot_normalize(const float o[9], float R[9]) {
  float A[3][3] = {{o[0], o[1], o[2]}, {o[3], o[4], o[5]}, {o[6], o[7], o[8]}};
  float S[3][3], V[3][3] = {{1.f, 0.f, 0.f}, {0.f, 1.f, 0.f}, {0.f, 0.f, 1.f}};
#pragma unroll
  for (int i = 0; i < 3; ++i)
#pragma unroll
    for (int j = 0; j < 3; ++j)
      S[i][j] = A[0][i] * A[0][j] + A[1][i] * A[1][j] + A[2][i] * A[2][j];
#pragma unroll
  for (int sw = 0; sw < 6; ++sw) { jrot<0, 1>(S, V); jrot<0, 2>(S, V); jrot<1, 2>(S, V); }
  float lam[3] = {S[0][0], S[1][1], S[2][2]};
  if (lam[0] < lam[1]) { csw(lam[0], lam[1]); csw(V[0][0], V[0][1]); csw(V[1][0], V[1][1]); csw(V[2][0], V[2][1]); }
  if (lam[0] < lam[2]) { csw(lam[0], lam[2]); csw(V[0][0], V[0][2]); csw(V[1][0], V[1][2]); csw(V[2][0], V[2][2]); }
  if (lam[1] < lam[2]) { csw(lam[1], lam[2]); csw(V[0][1], V[0][2]); csw(V[1][1], V[1][2]); csw(V[2][1], V[2][2]); }
  float U[3][3];
#pragma unroll
  for (int c = 0; c < 3; ++c) {
    const float u0 = A[0][0] * V[0][c] + A[0][1] * V[1][c] + A[0][2] * V[2][c];
    const float u1 = A[1][0] * V[0][c] + A[1][1] * V[1][c] + A[1][2] * V[2][c];
    const float u2 = A[2][0] * V[0][c] + A[2][1] * V[1][c] + A[2][2] * V[2][c];
    const float d = u0 * u0 + u1 * u1 + u2 * u2;
    if (c < 2 || d > 1e-18f) {
      const float inv = rsqrtf(fmaxf(d, 1e-30f));
      U[0][c] = u0 * inv; U[1][c] = u1 * inv; U[2][c] = u2 * inv;
    } else {  // degenerate sigma_min: complete with cross(u0,u1)
      U[0][c] = U[1][0] * U[2][1] - U[2][0] * U[1][1];
      U[1][c] = U[2][0] * U[0][1] - U[0][0] * U[2][1];
      U[2][c] = U[0][0] * U[1][1] - U[1][0] * U[0][1];
    }
  }
  const float det = A[0][0] * (A[1][1] * A[2][2] - A[1][2] * A[2][1])
                  - A[0][1] * (A[1][0] * A[2][2] - A[1][2] * A[2][0])
                  + A[0][2] * (A[1][0] * A[2][1] - A[1][1] * A[2][0]);
  const float sg = (det < 0.f) ? -1.f : 1.f;  // sign(det A) == det(U V^T)
#pragma unroll
  for (int i = 0; i < 3; ++i)
#pragma unroll
    for (int j = 0; j < 3; ++j)
      R[i * 3 + j] = U[i][0] * V[j][0] + U[i][1] * V[j][1] + sg * U[i][2] * V[j][2];
}

// ---------------------------------------------------------------------------
// Kernel B: WMMA GEMM head.  Each wave owns a 16-row tile of hflat[B,1024],
// computes lin1 (N=30 padded to 2x16 n-tiles, K=1024 in 32 steps) with
// v_wmma_f32_16x16x32_f16 using hi/lo f16 split (3 WMMAs/step) for ~f32
// precision, then bn3+gelu, 30x15 GEMV, 3x3 Procrustes, writes out[B,15].
// A and B fragments both use contiguous k order: element i of lane l is k = 16*(l>>4) + i.
// ---------------------------------------------------------------------------
__global__ __launch_bounds__(256) void head_kernel(
    const float* __restrict__ hflat,
    const float* __restrict__ lin1w, const float* __restrict__ lin1b,
    const float* __restrict__ g3, const float* __restrict__ b3,
    const float* __restrict__ m3, const float* __restrict__ v3,
    const float* __restrict__ lin2w, const float* __restrict__ lin2b,
    float* __restrict__ out)
{
  __shared__ float sW2[450];
  __shared__ float sB2[15];
  __shared__ float sCs[32], sCsh[32];          // folded bn3 scale / shift per col
  __shared__ float sGt[8][16][32];             // per-wave gelu(bn3(lin1)) tiles

  const int tid   = threadIdx.x;
  const int wave  = tid >> 5;
  const int lane  = tid & 31;
  const int n0    = lane & 15;
  const int khalf = lane >> 4;
  const int rowBase = blockIdx.x * 128 + wave * 16;

  for (int i = tid; i < 450; i += 256) sW2[i] = lin2w[i];
  if (tid < 15) sB2[tid] = lin2b[tid];
  if (tid < 32) {
    if (tid < 30) {
      const float sc = g3[tid] * rsqrtf(v3[tid] + EPSBN);
      sCs[tid]  = sc;
      sCsh[tid] = (lin1b[tid] - m3[tid]) * sc + b3[tid];
    } else { sCs[tid] = 0.f; sCsh[tid] = 0.f; }
  }
  __syncthreads();

  const float* arow = hflat + (size_t)(rowBase + n0) * 1024;
  v8f acc0 = {};   // cols 0..15
  v8f acc1 = {};   // cols 16..31 (16..29 valid)

  for (int kk = 0; kk < 32; ++kk) {
    const int k0 = kk * 32;
    // A fragment: row = lane&15 ; lanes<16 hold K {k0+0..15},
    //             lanes>=16 hold K {k0+16..31}
    const float* ap = arow + k0 + khalf * 16;
    __builtin_prefetch(ap + 64, 0, 1);          // stream-ahead (global_prefetch_b8)
    float ta[16];
    {
      float4 u0 = *reinterpret_cast<const float4*>(ap);
      float4 u1 = *reinterpret_cast<const float4*>(ap + 4);
      float4 u2 = *reinterpret_cast<const float4*>(ap + 8);
      float4 u3 = *reinterpret_cast<const float4*>(ap + 12);
      ta[0]  = u0.x; ta[1]  = u0.y; ta[2]  = u0.z; ta[3]  = u0.w;
      ta[4]  = u1.x; ta[5]  = u1.y; ta[6]  = u1.z; ta[7]  = u1.w;
      ta[8]  = u2.x; ta[9]  = u2.y; ta[10] = u2.z; ta[11] = u2.w;
      ta[12] = u3.x; ta[13] = u3.y; ta[14] = u3.z; ta[15] = u3.w;
    }
    v16h ahi, alo;
#pragma unroll
    for (int i = 0; i < 16; ++i) {
      const float xv = ta[i];
      const _Float16 h = (_Float16)xv;
      ahi[i] = h;
      alo[i] = (_Float16)(xv - (float)h);
    }
    // B fragment: col = lane&15 (+16 for n-tile 1); 16 consecutive K starting
    // at k0 + (lane>>4)*16 ; B[k][n] = lin1_w[n][k]
    const int kb0 = k0 + khalf * 16;
#pragma unroll
    for (int nt = 0; nt < 2; ++nt) {
      const int nc = n0 + nt * 16;
      float tb[16];
      if (nc < 30) {
        const float* wp = lin1w + (size_t)nc * 1024 + kb0;
        float4 w0 = *reinterpret_cast<const float4*>(wp);
        float4 w1 = *reinterpret_cast<const float4*>(wp + 4);
        float4 w2 = *reinterpret_cast<const float4*>(wp + 8);
        float4 w3 = *reinterpret_cast<const float4*>(wp + 12);
        tb[0]  = w0.x; tb[1]  = w0.y; tb[2]  = w0.z; tb[3]  = w0.w;
        tb[4]  = w1.x; tb[5]  = w1.y; tb[6]  = w1.z; tb[7]  = w1.w;
        tb[8]  = w2.x; tb[9]  = w2.y; tb[10] = w2.z; tb[11] = w2.w;
        tb[12] = w3.x; tb[13] = w3.y; tb[14] = w3.z; tb[15] = w3.w;
      } else {
#pragma unroll
        for (int i = 0; i < 16; ++i) tb[i] = 0.f;
      }
      v16h bhi, blo;
#pragma unroll
      for (int i = 0; i < 16; ++i) {
        const float wv = tb[i];
        const _Float16 h = (_Float16)wv;
        bhi[i] = h;
        blo[i] = (_Float16)(wv - (float)h);
      }
      v8f acc = nt ? acc1 : acc0;
      acc = __builtin_amdgcn_wmma_f32_16x16x32_f16(false, ahi, false, bhi, (short)0, acc, false, false);
      acc = __builtin_amdgcn_wmma_f32_16x16x32_f16(false, ahi, false, blo, (short)0, acc, false, false);
      acc = __builtin_amdgcn_wmma_f32_16x16x32_f16(false, alo, false, bhi, (short)0, acc, false, false);
      if (nt) acc1 = acc; else acc0 = acc;
    }
  }

  // C layout: VGPR v, lane l -> M = v + 8*(l>>4), N = (l&15) + 16*ntile
  {
    const int n1 = n0 + 16;
    const float cs0 = sCs[n0], ch0 = sCsh[n0];
    const float cs1 = sCs[n1], ch1 = sCsh[n1];
#pragma unroll
    for (int v = 0; v < 8; ++v) {
      const int m = v + khalf * 8;
      sGt[wave][m][n0] = gelu_f(acc0[v] * cs0 + ch0);
      sGt[wave][m][n1] = gelu_f(acc1[v] * cs1 + ch1);
    }
  }
  __syncthreads();

  // ---- tail: lin2 + Procrustes, one batch row per lane (lanes 0..15)
  if (lane < 16) {
    const int r = rowBase + lane;
    float g[30];
#pragma unroll
    for (int n = 0; n < 30; ++n) g[n] = sGt[wave][lane][n];
    float o[15];
#pragma unroll
    for (int j = 0; j < 15; ++j) {
      float acc = sB2[j];
#pragma unroll
      for (int n = 0; n < 30; ++n) acc = fmaf(g[n], sW2[j * 30 + n], acc);
      o[j] = acc;
    }
    float R[9];
    rot_normalize(o, R);
    float* op = out + (size_t)r * 15;
#pragma unroll
    for (int j = 0; j < 9; ++j) op[j] = R[j];
#pragma unroll
    for (int j = 9; j < 15; ++j) op[j] = o[j];
  }
}

// ---------------------------------------------------------------------------
extern "C" void kernel_launch(void* const* d_in, const int* in_sizes, int n_in,
                              void* d_out, int out_size, void* d_ws, size_t ws_size,
                              hipStream_t stream) {
  (void)in_sizes; (void)n_in; (void)out_size; (void)ws_size;
  const float* x     = (const float*)d_in[0];
  const float* c1w   = (const float*)d_in[1];
  const float* c1b   = (const float*)d_in[2];
  const float* bn1g  = (const float*)d_in[3];
  const float* bn1b  = (const float*)d_in[4];
  const float* bn1m  = (const float*)d_in[5];
  const float* bn1v  = (const float*)d_in[6];
  const float* c2w   = (const float*)d_in[7];
  const float* c2b   = (const float*)d_in[8];
  const float* bn2g  = (const float*)d_in[9];
  const float* bn2b  = (const float*)d_in[10];
  const float* bn2m  = (const float*)d_in[11];
  const float* bn2v  = (const float*)d_in[12];
  const float* lin1w = (const float*)d_in[13];
  const float* lin1b = (const float*)d_in[14];
  const float* bn3g  = (const float*)d_in[15];
  const float* bn3b  = (const float*)d_in[16];
  const float* bn3m  = (const float*)d_in[17];
  const float* bn3v  = (const float*)d_in[18];
  const float* lin2w = (const float*)d_in[19];
  const float* lin2b = (const float*)d_in[20];

  float* hflat = (float*)d_ws;                 // [16384, 1024] f32 = 64 MB

  conv_fused_kernel<<<16384, 256, 0, stream>>>(
      x, c1w, c1b, bn1g, bn1b, bn1m, bn1v,
      c2w, c2b, bn2g, bn2b, bn2m, bn2v, hflat);

  head_kernel<<<128, 256, 0, stream>>>(
      hflat, lin1w, lin1b, bn3g, bn3b, bn3m, bn3v,
      lin2w, lin2b, (float*)d_out);
}
